// EpisodicRetrieval_72164040507982
// MI455X (gfx1250) — matrix operand order repaired and checked by reading, not compiled, not run on hardware
//
#include <hip/hip_runtime.h>
#include <hip/hip_bf16.h>

// ---------------------------------------------------------------------------
// EpisodicRetrieval for MI455X (gfx1250, wave32, WMMA).
//
// Roofline: the 256x100000x256 distance GEMM = 13.1 GFLOP over 51MB (bf16) of
// streamed memory. This version reads `memory` from HBM exactly once:
// each block owns a 25-tile column stripe, keeps all 256 query rows of A
// resident in VGPRs (2 row-tiles per wave), and stages each 16-row B tile in
// LDS (async-to-LDS) shared by all 8 waves. That puts the kernel at the
// 23.3TB/s HBM roofline (~2-3us) instead of 16x over it. Distances only feed
// a top-16 ranking, so bf16 dot + exact fp32 norms is precision-safe.
// Value-carrying k/v projections use full-precision fp32 WMMA
// (v_wmma_f32_16x16x4_f32, verified to lower); tiny GEMMs/softmax stay VALU.
// ---------------------------------------------------------------------------

typedef __bf16 v16bf __attribute__((ext_vector_type(16)));
typedef __bf16 v8bf  __attribute__((ext_vector_type(8)));
typedef float  v8f   __attribute__((ext_vector_type(8)));
typedef float  v2f   __attribute__((ext_vector_type(2)));
typedef int    v4i_  __attribute__((vector_size(16)));   // matches builtin param

#define FMAX_ 3.402823466e+38f

// Problem constants
#define B_    256
#define N_    100000
#define D_    256
#define K_    16
// 100000/16 = 6250 column tiles = 250 blocks x 25 tiles (exact, no bounds).
#define NBLK_       250
#define TILES_PB    25
#define NCAND       (NBLK_ * K_)    // 4000 candidates per query row
#define BPAD        264             // B tile LDS row stride (bf16 elems), pad 8
#define SPAD        17              // strip LDS row stride (floats)

#if __has_builtin(__builtin_amdgcn_global_load_async_to_lds_b128)
#define HAVE_ASYNC_LDS 1
#endif

typedef __attribute__((address_space(1))) v4i_* gptr4_t;
typedef __attribute__((address_space(3))) v4i_* lptr4_t;

__device__ __forceinline__ void async_b128(const void* g, void* l) {
#ifdef HAVE_ASYNC_LDS
    __builtin_amdgcn_global_load_async_to_lds_b128(
        (gptr4_t)(uintptr_t)g, (lptr4_t)(uintptr_t)l, 0, 0);
#else
    *(float4*)l = *(const float4*)g;
#endif
}

__device__ __forceinline__ void wait_async() {
#ifdef HAVE_ASYNC_LDS
#if __has_builtin(__builtin_amdgcn_s_wait_asynccnt)
    __builtin_amdgcn_s_wait_asynccnt(0);
#else
    asm volatile("s_wait_asynccnt 0x0" ::: "memory");
#endif
#endif
}

// ---------------------------------------------------------------------------
// 1) fp32 row -> bf16 row + exact fp32 squared norm. One block per row.
// ---------------------------------------------------------------------------
__global__ void __launch_bounds__(256) prep_row_kernel(
    const float* __restrict__ src, unsigned short* __restrict__ dstBf,
    float* __restrict__ norms)
{
    const int row = blockIdx.x, t = threadIdx.x;
    const float v = src[(size_t)row * D_ + t];
    __bf16 h = (__bf16)v;
    dstBf[(size_t)row * D_ + t] = __builtin_bit_cast(unsigned short, h);
    __shared__ float red[256];
    red[t] = v * v;
    __syncthreads();
    #pragma unroll
    for (int off = 128; off > 0; off >>= 1) {
        if (t < off) red[t] += red[t + off];
        __syncthreads();
    }
    if (t == 0) norms[row] = red[0];
}

// ---------------------------------------------------------------------------
// 2) Distance GEMM (bf16 WMMA, fp32 accum) fused with per-row top-K.
//    grid = 250 blocks x 256 threads (8 waves). Block owns 25 column tiles.
//    Wave w computes row-tiles 2w and 2w+1 (rows 32w..32w+31); A for both
//    tiles is loaded once into VGPRs. Per column tile, the 16x256 bf16 B tile
//    is staged in LDS (async copy) and shared by all waves -> memory matrix
//    streamed from HBM exactly once.
//
//    A (16x32 bf16) lanes 0-15: M=lane, K=0..7 / 16..23; lanes 16-31:
//    K=8..15 / 24..31.  B (32x16): lane = N mod 16; lanes 0-15 K=0..7 / 16..23,
//    lanes 16-31 K=8..15 / 24..31 (same order as A).  C/D (16x16 f32): lanes 0-15
//    N=lane M=0..7 in acc[0..7]; lanes 16-31 N=lane-16 M=8..15.
// ---------------------------------------------------------------------------
__global__ void __launch_bounds__(256) dist_topk_kernel(
    const unsigned short* __restrict__ pBfU,
    const unsigned short* __restrict__ memBfU,
    const float* __restrict__ pnorm, const float* __restrict__ mnorm,
    float* __restrict__ candV, int* __restrict__ candI)
{
    __shared__ __bf16 s_b[16 * BPAD];        // staged B tile
    __shared__ float  s_strip[256 * SPAD];   // 256 rows x 16 cols of d^2
    __shared__ float  s_pn[256];

    const int t = threadIdx.x;
    const int wave = t >> 5, lane = t & 31;
    const int blk = blockIdx.x;
    const int laneHalf = (lane >= 16) ? 1 : 0;
    const int laneM = lane & 15;

    s_pn[t] = pnorm[t];

    // ---- load both A row-tiles for this wave into registers (held live) ----
    const __bf16* __restrict__ pBf = (const __bf16*)pBfU;
    const __bf16* prow0 = pBf + (size_t)(32 * wave + laneM) * D_;
    const __bf16* prow1 = prow0 + (size_t)16 * D_;
    v16bf A0[8], A1[8];
    #pragma unroll
    for (int ks = 0; ks < 8; ++ks) {
        const int aOff = ks * 32 + laneHalf * 8;
        v8bf lo0 = *(const v8bf*)(prow0 + aOff);
        v8bf hi0 = *(const v8bf*)(prow0 + aOff + 16);
        A0[ks] = __builtin_shufflevector(lo0, hi0,
            0, 1, 2, 3, 4, 5, 6, 7, 8, 9, 10, 11, 12, 13, 14, 15);
        v8bf lo1 = *(const v8bf*)(prow1 + aOff);
        v8bf hi1 = *(const v8bf*)(prow1 + aOff + 16);
        A1[ks] = __builtin_shufflevector(lo1, hi1,
            0, 1, 2, 3, 4, 5, 6, 7, 8, 9, 10, 11, 12, 13, 14, 15);
    }

    // per-thread (per query row t) sorted ascending top-16, static indexing
    float tv[16]; int ti[16];
    #pragma unroll
    for (int s = 0; s < 16; ++s) { tv[s] = FMAX_; ti[s] = -1; }

    const unsigned short* __restrict__ memBf = memBfU;
    // B staging role of this thread: row t>>4 of the tile, 32B chunk t&15
    const int bRow = t >> 4;
    const int bOff = (t & 15) * 16;           // bf16 elems (32B)
    __bf16* ldsBdst = s_b + bRow * BPAD + bOff;

    for (int it = 0; it < TILES_PB; ++it) {
        const int tileCol = blk * TILES_PB + it;
        const int colBase = tileCol * 16;

        __syncthreads();   // strip reads + B reads of previous iter complete
        {
            const unsigned short* g =
                memBf + (size_t)(colBase + bRow) * D_ + bOff;
            async_b128(g, ldsBdst);
            async_b128(g + 8, ldsBdst + 8);
            if (it + 1 < TILES_PB)   // prefetch next tile's chunk
                __builtin_prefetch((const void*)(g + D_ * 16), 0, 1);
        }
        wait_async();
        __syncthreads();   // B tile visible to all waves

        const float mn = mnorm[colBase + laneM];

        v8f acc0 = {}, acc1 = {};
        #pragma unroll
        for (int ks = 0; ks < 8; ++ks) {
            v8bf blo = *(const v8bf*)(s_b + laneM * BPAD + ks * 32 + laneHalf * 8);
            v8bf bhi = *(const v8bf*)(s_b + laneM * BPAD + ks * 32 + laneHalf * 8 + 16);
            v16bf b = __builtin_shufflevector(blo, bhi,
                0, 1, 2, 3, 4, 5, 6, 7, 8, 9, 10, 11, 12, 13, 14, 15);
            acc0 = __builtin_amdgcn_wmma_f32_16x16x32_bf16(
                false, A0[ks], false, b, (short)0, acc0, false, false);
            acc1 = __builtin_amdgcn_wmma_f32_16x16x32_bf16(
                false, A1[ks], false, b, (short)0, acc1, false, false);
        }

        #pragma unroll
        for (int i = 0; i < 8; ++i) {
            const int row0 = 32 * wave + laneHalf * 8 + i;
            const int row1 = row0 + 16;
            s_strip[row0 * SPAD + laneM] = s_pn[row0] + mn - 2.0f * acc0[i];
            s_strip[row1 * SPAD + laneM] = s_pn[row1] + mn - 2.0f * acc1[i];
        }
        __syncthreads();   // strip visible

        // thread t scans its own query row over these 16 columns
        #pragma unroll
        for (int j = 0; j < 16; ++j) {
            const float v = s_strip[t * SPAD + j];
            if (v < tv[15]) {
                tv[15] = v; ti[15] = colBase + j;
                #pragma unroll
                for (int s = 15; s > 0; --s) {
                    if (tv[s] < tv[s - 1]) {
                        float fv = tv[s]; tv[s] = tv[s - 1]; tv[s - 1] = fv;
                        int   iv = ti[s]; ti[s] = ti[s - 1]; ti[s - 1] = iv;
                    }
                }
            }
        }
    }

    // thread t owns query row t: write its 16 candidates for this block
    float* cvOut = candV + (size_t)t * NCAND + blk * K_;
    int*   ciOut = candI + (size_t)t * NCAND + blk * K_;
    #pragma unroll
    for (int s = 0; s < K_; ++s) { cvOut[s] = tv[s]; ciOut[s] = ti[s]; }
}

// ---------------------------------------------------------------------------
// 3) Merge NCAND=4000 candidates per row into the final 16 (order irrelevant:
//    softmax attention over K is permutation invariant). One block per row.
// ---------------------------------------------------------------------------
__global__ void __launch_bounds__(256) topk_merge_kernel(
    const float* __restrict__ candV, const int* __restrict__ candI,
    int* __restrict__ topkIdx)
{
    __shared__ float s_v[NCAND];
    __shared__ int   s_i[NCAND];
    __shared__ float r_v[256];
    __shared__ int   r_p[256];
    const int b = blockIdx.x, t = threadIdx.x;
    for (int e = t; e < NCAND; e += 256) {
        s_v[e] = candV[(size_t)b * NCAND + e];
        s_i[e] = candI[(size_t)b * NCAND + e];
    }
    __syncthreads();
    for (int k = 0; k < K_; ++k) {
        float best = FMAX_; int bp = -1;
        for (int e = t; e < NCAND; e += 256) {
            if (s_v[e] < best) { best = s_v[e]; bp = e; }
        }
        r_v[t] = best; r_p[t] = bp;
        __syncthreads();
        for (int off = 128; off > 0; off >>= 1) {
            if (t < off && r_v[t + off] < r_v[t]) {
                r_v[t] = r_v[t + off]; r_p[t] = r_p[t + off];
            }
            __syncthreads();
        }
        if (t == 0) {
            const int p = r_p[0];
            int idx = (p >= 0) ? s_i[p] : 0;
            topkIdx[b * K_ + k] = (idx >= 0) ? idx : 0;
            if (p >= 0) s_v[p] = FMAX_;
        }
        __syncthreads();
    }
}

// ---------------------------------------------------------------------------
// 4) Gather retrieved rows (fp32). One block (64 threads, float4) per row.
// ---------------------------------------------------------------------------
__global__ void __launch_bounds__(64) gather_kernel(
    const float* __restrict__ memory, const int* __restrict__ topkIdx,
    float* __restrict__ retr)
{
    const int r = blockIdx.x;                 // 0..B*K-1
    const int idx = topkIdx[r];
    const float4* s = (const float4*)(memory + (size_t)idx * D_);
    float4* d = (float4*)(retr + (size_t)r * D_);
    d[threadIdx.x] = s[threadIdx.x];
}

// ---------------------------------------------------------------------------
// 5) k/v projections in full-precision fp32 WMMA (v_wmma_f32_16x16x4_f32).
//    A (16x4 f32): all lanes M = lane&15; lanes 0-15 K=0,1; lanes 16-31 K=2,3.
//    B (4x16): lane = N mod 16, same K split. C/D layout as above.
//    grid = (512, 2): 8 waves/block * 512 = 4096 tiles; y picks K vs V.
// ---------------------------------------------------------------------------
__global__ void __launch_bounds__(256) kv_proj_kernel(
    const float* __restrict__ retr,
    const float* __restrict__ wk, const float* __restrict__ bk,
    const float* __restrict__ wv, const float* __restrict__ bv,
    float* __restrict__ kout, float* __restrict__ vout)
{
    const float* W; const float* bias; float* out;
    if (blockIdx.y == 0) { W = wk; bias = bk; out = kout; }
    else                 { W = wv; bias = bv; out = vout; }

    const int t = threadIdx.x, wave = t >> 5, lane = t & 31;
    const int tile = blockIdx.x * 8 + wave;     // 0..4095
    const int mTile = tile >> 4, nTile = tile & 15;
    const int laneHalf = (lane >= 16) ? 1 : 0;
    const int laneN = lane & 15;
    const int m0 = mTile * 16, n0 = nTile * 16;
    const int n = n0 + laneN;

    v8f acc = {};
#if __has_builtin(__builtin_amdgcn_wmma_f32_16x16x4_f32)
    const float* arow = retr + (size_t)(m0 + laneN) * D_;
    const float* brow = W + (size_t)n * D_;
    #pragma unroll
    for (int ks = 0; ks < 64; ++ks) {
        const int koff = ks * 4 + (laneHalf ? 2 : 0);
        v2f a  = *(const v2f*)(arow + koff);
        v2f bb = *(const v2f*)(brow + koff);
        acc = __builtin_amdgcn_wmma_f32_16x16x4_f32(
            false, a, false, bb, (short)0, acc, false, false);
    }
#else
    const float* brow = W + (size_t)n * D_;
    #pragma unroll
    for (int i = 0; i < 8; ++i) {
        const float* ar = retr + (size_t)(m0 + laneHalf * 8 + i) * D_;
        float s = 0.f;
        for (int j = 0; j < D_; ++j) s = fmaf(ar[j], brow[j], s);
        acc[i] = s;
    }
#endif
    const float bn = bias[n];
    #pragma unroll
    for (int i = 0; i < 8; ++i) {
        const int m = m0 + laneHalf * 8 + i;
        out[(size_t)m * D_ + n] = acc[i] + bn;
    }
}

// ---------------------------------------------------------------------------
// 6) Attention: scores (1/8 scale), softmax over K=16, context. Block per b.
// ---------------------------------------------------------------------------
__global__ void __launch_bounds__(256) attention_kernel(
    const float* __restrict__ qbuf, const float* __restrict__ kbuf,
    const float* __restrict__ vbuf, float* __restrict__ ctxBuf)
{
    __shared__ float s_q[256];
    __shared__ float s_sc[64];
    const int b = blockIdx.x, t = threadIdx.x;
    s_q[t] = qbuf[(size_t)b * D_ + t];
    __syncthreads();
    if (t < 64) {
        const int h = t >> 4, k = t & 15;
        const float* krow = kbuf + ((size_t)(b * K_ + k)) * D_ + h * 64;
        const float* qh = s_q + h * 64;
        float s = 0.f;
        #pragma unroll 8
        for (int d = 0; d < 64; ++d) s = fmaf(qh[d], krow[d], s);
        s_sc[t] = s * 0.125f;          // 1/sqrt(64)
    }
    __syncthreads();
    if (t < 4) {
        float mx = -FMAX_;
        for (int k = 0; k < 16; ++k) mx = fmaxf(mx, s_sc[t * 16 + k]);
        float sum = 0.f;
        for (int k = 0; k < 16; ++k) {
            const float e = __expf(s_sc[t * 16 + k] - mx);
            s_sc[t * 16 + k] = e; sum += e;
        }
        const float inv = 1.f / sum;
        for (int k = 0; k < 16; ++k) s_sc[t * 16 + k] *= inv;
    }
    __syncthreads();
    const int h = t >> 6;
    float c = 0.f;
    #pragma unroll
    for (int k = 0; k < 16; ++k)
        c = fmaf(s_sc[h * 16 + k],
                 vbuf[((size_t)(b * K_ + k)) * D_ + t], c);
    ctxBuf[(size_t)b * D_ + t] = c;
}

// ---------------------------------------------------------------------------
// 7) GEMV: Y[b][o] = X[b][:] . W[o][:] + bias[o] (opt ReLU). Block per b.
// ---------------------------------------------------------------------------
__global__ void __launch_bounds__(256) gemv256_kernel(
    const float* __restrict__ X, const float* __restrict__ W,
    const float* __restrict__ bias, float* __restrict__ Y, int relu)
{
    __shared__ float sx[256];
    const int b = blockIdx.x, t = threadIdx.x;
    sx[t] = X[(size_t)b * D_ + t];
    __syncthreads();
    const float* wrow = W + (size_t)t * D_;
    float acc = 0.f;
    #pragma unroll 8
    for (int j = 0; j < D_; ++j) acc = fmaf(sx[j], wrow[j], acc);
    acc += bias[t];
    if (relu) acc = fmaxf(acc, 0.f);
    Y[(size_t)b * D_ + t] = acc;
}

// ---------------------------------------------------------------------------
// Launcher
// ---------------------------------------------------------------------------
extern "C" void kernel_launch(void* const* d_in, const int* in_sizes, int n_in,
                              void* d_out, int out_size, void* d_ws, size_t ws_size,
                              hipStream_t stream) {
    (void)in_sizes; (void)n_in; (void)out_size; (void)ws_size;
    const float* p_t    = (const float*)d_in[0];
    const float* memory = (const float*)d_in[1];
    const float* wq = (const float*)d_in[2];  const float* bq = (const float*)d_in[3];
    const float* wk = (const float*)d_in[4];  const float* bk = (const float*)d_in[5];
    const float* wv = (const float*)d_in[6];  const float* bv = (const float*)d_in[7];
    const float* wo = (const float*)d_in[8];  const float* bo = (const float*)d_in[9];
    const float* w1 = (const float*)d_in[10]; const float* b1 = (const float*)d_in[11];
    const float* w2 = (const float*)d_in[12]; const float* b2 = (const float*)d_in[13];
    float* out = (float*)d_out;

    // carve workspace (~73 MB total)
    char* ws = (char*)d_ws;
    size_t off = 0;
    auto take = [&](size_t bytes) -> char* {
        char* p = ws + off;
        off = (off + bytes + 255) & ~(size_t)255;
        return p;
    };
    unsigned short* memBf = (unsigned short*)take((size_t)N_ * D_ * 2);
    float* mnorm  = (float*)take((size_t)N_ * 4);
    unsigned short* pBf = (unsigned short*)take((size_t)B_ * D_ * 2);
    float* pnorm  = (float*)take((size_t)B_ * 4);
    float* candV  = (float*)take((size_t)B_ * NCAND * 4);
    int*   candI  = (int*)  take((size_t)B_ * NCAND * 4);
    int*   topkIdx= (int*)  take((size_t)B_ * K_ * 4);
    float* retr   = (float*)take((size_t)B_ * K_ * D_ * 4);
    float* qbuf   = (float*)take((size_t)B_ * D_ * 4);
    float* kbuf   = (float*)take((size_t)B_ * K_ * D_ * 4);
    float* vbuf   = (float*)take((size_t)B_ * K_ * D_ * 4);
    float* ctxBuf = (float*)take((size_t)B_ * D_ * 4);
    float* attBuf = (float*)take((size_t)B_ * D_ * 4);
    float* h1buf  = (float*)take((size_t)B_ * D_ * 4);

    prep_row_kernel<<<N_, 256, 0, stream>>>(memory, memBf, mnorm);
    prep_row_kernel<<<B_, 256, 0, stream>>>(p_t, pBf, pnorm);
    gemv256_kernel<<<B_, 256, 0, stream>>>(p_t, wq, bq, qbuf, 0);
    dist_topk_kernel<<<NBLK_, 256, 0, stream>>>(
        pBf, memBf, pnorm, mnorm, candV, candI);
    topk_merge_kernel<<<B_, 256, 0, stream>>>(candV, candI, topkIdx);
    gather_kernel<<<B_ * K_, 64, 0, stream>>>(memory, topkIdx, retr);
    kv_proj_kernel<<<dim3(512, 2), 256, 0, stream>>>(
        retr, wk, bk, wv, bv, kbuf, vbuf);
    attention_kernel<<<B_, 256, 0, stream>>>(qbuf, kbuf, vbuf, ctxBuf);
    gemv256_kernel<<<B_, 256, 0, stream>>>(ctxBuf, wo, bo, attBuf, 0);
    gemv256_kernel<<<B_, 256, 0, stream>>>(attBuf, w1, b1, h1buf, 1);
    gemv256_kernel<<<B_, 256, 0, stream>>>(h1buf, w2, b2, out, 0);
}
